// ImMatchNet_15719580303922
// MI455X (gfx1250) — matrix operand order repaired and checked by reading, not compiled, not run on hardware
//
#include <hip/hip_runtime.h>
#include <hip/hip_bf16.h>

typedef __attribute__((ext_vector_type(8)))  __bf16 v8bf;
typedef __attribute__((ext_vector_type(16))) __bf16 v16bf;
typedef __attribute__((ext_vector_type(8)))  float  v8f;
typedef __attribute__((ext_vector_type(4)))  unsigned int v4u;
typedef __attribute__((ext_vector_type(8)))  int v8i;
typedef __attribute__((ext_vector_type(4)))  int v4i;

#define FD   32                       // each of f1..f4
#define NPOS 1024                     // f1*f2 == f3*f4
#define POS4 (1048576)                // 32^4
#define NB   4                        // batch

__device__ __forceinline__ __bf16 f2bf(float f) {
    union { float f; unsigned u; } v; v.f = f;
    unsigned r = v.u + 0x7FFFu + ((v.u >> 16) & 1u);   // round-to-nearest-even
    unsigned short h = (unsigned short)(r >> 16);
    return __builtin_bit_cast(__bf16, h);
}

// ---------------------------------------------------------------------------
// 1) L2-normalize over channels, write transposed bf16 [b][pos][c]
// ---------------------------------------------------------------------------
__global__ void l2norm_transpose_kernel(const float* __restrict__ f,
                                        __bf16* __restrict__ out,
                                        int cdim, int n) {
    int idx = blockIdx.x * blockDim.x + threadIdx.x;     // over NB*n
    if (idx >= NB * n) return;
    int b = idx / n, p = idx % n;
    const float* src = f + (size_t)b * cdim * n + p;
    float s = 0.f;
    for (int c = 0; c < cdim; ++c) { float v = src[(size_t)c * n]; s += v * v; }
    float inv = 1.0f / sqrtf(s + 1e-6f);
    __bf16* dst = out + ((size_t)b * n + p) * cdim;
    for (int c = 0; c < cdim; ++c) dst[c] = f2bf(src[(size_t)c * n] * inv);
}

// ---------------------------------------------------------------------------
// 2) Correlation GEMM via bf16 WMMA, relu + channel l2norm epilogue.
//    One wave per 16x16 tile, 8 waves per block.
// ---------------------------------------------------------------------------
__global__ void __launch_bounds__(256)
corr_wmma_kernel(const __bf16* __restrict__ faT,     // [NB][1024][256]
                 const __bf16* __restrict__ fbT,     // [NB][1024][256]
                 float* __restrict__ corr) {         // [NB][1024][1024]
    int lane = threadIdx.x & 31, wave = threadIdx.x >> 5;
    int tile = blockIdx.x * 8 + wave;                // NB*64*64 = 16384 tiles
    int b  = tile >> 12;
    int tr = tile & 4095;
    int ti = tr >> 6, tj = tr & 63;
    int half = lane >> 4, lm = lane & 15;

    const __bf16* arow = faT + ((size_t)b * NPOS + ti * 16 + lm) * 256;
    const __bf16* brow = fbT + ((size_t)b * NPOS + tj * 16 + lm) * 256;

    v8f acc = {};
    for (int k = 0; k < 256; k += 32) {
        __builtin_prefetch(arow + k + 64, 0, 0);
        __builtin_prefetch(brow + k + 64, 0, 0);
        // A (16x32, MxK): lanes 0-15 hold K [k..k+16); lanes 16-31: K [k+16..k+32)
        v16bf a = *(const v16bf*)(arow + k + half * 16);
        // B (32x16, KxN): lane L holds column N=L, K [k..k+16); lane L+16: K [k+16..k+32)
        v16bf bv = *(const v16bf*)(brow + k + half * 16);
        acc = __builtin_amdgcn_wmma_f32_16x16x32_bf16(
                  false, a, false, bv, (short)0, acc, false, false);
    }
    #pragma unroll
    for (int r = 0; r < 8; ++r) {
        int i = ti * 16 + r + half * 8;
        int j = tj * 16 + lm;
        float v = acc[r];
        v = v > 0.f ? v : 0.f;
        v = v / sqrtf(v * v + 1e-6f);
        corr[((size_t)b << 20) + ((size_t)i << 10) + j] = v;
    }
}

// ---------------------------------------------------------------------------
// 3) Max reductions for mutual matching
// ---------------------------------------------------------------------------
__global__ void rowmax_kernel(const float* __restrict__ m,
                              float* __restrict__ outmax, int cols) {
    int row = blockIdx.x;                             // NB*1024 rows
    const float* p = m + (size_t)row * cols;
    float v = -1e30f;
    for (int j = threadIdx.x; j < cols; j += blockDim.x) v = fmaxf(v, p[j]);
    __shared__ float s[256];
    s[threadIdx.x] = v; __syncthreads();
    for (int off = 128; off > 0; off >>= 1) {
        if ((int)threadIdx.x < off) s[threadIdx.x] = fmaxf(s[threadIdx.x], s[threadIdx.x + off]);
        __syncthreads();
    }
    if (threadIdx.x == 0) outmax[row] = s[0];
}

__global__ void colmax_kernel(const float* __restrict__ m,
                              float* __restrict__ outmax) {
    int idx = blockIdx.x * blockDim.x + threadIdx.x;  // NB*1024
    if (idx >= NB * NPOS) return;
    int b = idx >> 10, j = idx & 1023;
    const float* p = m + ((size_t)b << 20) + j;
    float v = -1e30f;
    for (int i = 0; i < NPOS; ++i) v = fmaxf(v, p[(size_t)i << 10]);
    outmax[idx] = v;
}

// ---------------------------------------------------------------------------
// 4) Mutual matching: c * c/(maxA[b,i]+eps) * c/(maxB[b,j]+eps)
// ---------------------------------------------------------------------------
__global__ void mutual_match_kernel(const float* __restrict__ c,
                                    const float* __restrict__ maxA,
                                    const float* __restrict__ maxB,
                                    float* __restrict__ out) {
    size_t idx = (size_t)blockIdx.x * blockDim.x + threadIdx.x;   // NB<<20
    if (idx >= ((size_t)NB << 20)) return;
    int b = (int)(idx >> 20);
    int i = (int)((idx >> 10) & 1023);
    int j = (int)(idx & 1023);
    float v = c[idx];
    out[idx] = v * (v / (maxA[(b << 10) + i] + 1e-5f))
                 * (v / (maxB[(b << 10) + j] + 1e-5f));
}

// ---------------------------------------------------------------------------
// 5) Per-batch 1024x1024 transpose ((f1,f2) <-> (f3,f4))
// ---------------------------------------------------------------------------
__global__ void transpose_kernel(const float* __restrict__ in,
                                 float* __restrict__ out) {
    size_t idx = (size_t)blockIdx.x * blockDim.x + threadIdx.x;
    if (idx >= ((size_t)NB << 20)) return;
    int b = (int)(idx >> 20), i = (int)((idx >> 10) & 1023), j = (int)(idx & 1023);
    out[((size_t)b << 20) + ((size_t)j << 10) + i] = in[idx];
}

// ---------------------------------------------------------------------------
// 6) Conv4d (3^4 taps) as implicit GEMM via bf16 WMMA.
//    Weights DMA'd to LDS once per block by the Tensor Data Mover
//    (tensor_load_to_lds, TENSORcnt), then packed as a row-major bf16 [16][K]
//    A matrix (81 taps zero-padded to 96 so 32-wide K chunks never straddle
//    a channel). B = im2col gather of 16 f4-consecutive positions.
// ---------------------------------------------------------------------------
__global__ void __launch_bounds__(256)
conv4d_wmma_kernel(const float* __restrict__ x,      // [NB, Ci, 32^4]
                   const float* __restrict__ w,      // [Co, Ci, 81]
                   const float* __restrict__ bias,   // [Co]
                   float* __restrict__ y,            // [NB, Co, 32^4]
                   int Ci, int Co) {
    const int K = Ci * 96;
    extern __shared__ char smem[];
    __bf16* aLds = (__bf16*)smem;                           // [16][K] packed A
    const unsigned rawOff = (unsigned)((16 * K * 2 + 15) & ~15);
    float* wRaw = (float*)(smem + rawOff);                  // [Co*Ci*81] raw f32

    // --- TDM: DMA the whole weight blob into LDS (wave 0 issues, all wait) ---
    const unsigned nW = (unsigned)(Co * Ci * 81);           // dwords
    if ((threadIdx.x >> 5) == 0) {
        unsigned long long ga = (unsigned long long)(uintptr_t)w;
        unsigned ldsAddr = (unsigned)(uintptr_t)(void*)wRaw; // LDS byte offset
        v4u g0;
        g0.x = 1u;                                           // count=1, user mode
        g0.y = ldsAddr;                                      // lds_addr
        g0.z = (unsigned)(ga & 0xFFFFFFFFu);                 // global_addr[31:0]
        g0.w = (unsigned)((ga >> 32) & 0x01FFFFFFu)          // global_addr[56:32]
             | (2u << 30);                                   // type = 2 (image)
        v8i g1 = {};
        g1[0] = (int)(2u << 16);                             // data_size = 2 (4B)
        g1[1] = (int)((nW & 0xFFFFu) << 16);                 // tensor_dim0[15:0]
        g1[2] = (int)((nW >> 16) & 0xFFFFu)                  // tensor_dim0[31:16]
              | (int)(1u << 16);                             // tensor_dim1 = 1
        g1[3] = (int)((nW & 0xFFFFu) << 16);                 // tile_dim0 = nW
        g1[4] = 1;                                           // tile_dim1=1, tile_dim2=0
        g1[5] = (int)nW;                                     // tensor_dim0_stride[31:0]
        v4i z4 = {};
#if defined(__clang_major__) && (__clang_major__ >= 23)
        v8i z8 = {};
        __builtin_amdgcn_tensor_load_to_lds(g0, g1, z4, z4, z8, 0);
#else
        __builtin_amdgcn_tensor_load_to_lds(g0, g1, z4, z4, 0);
#endif
        __builtin_amdgcn_s_wait_tensorcnt(0);
    }
    __syncthreads();

    // --- pack bf16 A-matrix (zero pad: m>=Co, tap in [81,96)) from LDS ---
    for (int t = threadIdx.x; t < 16 * K; t += blockDim.x) {
        int m = t / K, k = t % K;
        int ci = k / 96, tap = k % 96;
        __bf16 v = f2bf(0.0f);
        if (m < Co && tap < 81) v = f2bf(wRaw[(m * Ci + ci) * 81 + tap]);
        aLds[t] = v;
    }
    __syncthreads();

    int lane = threadIdx.x & 31, wave = threadIdx.x >> 5;
    long tile = (long)blockIdx.x * 8 + wave;         // NB*32*32*32*2 tiles
    int f4t = (int)(tile & 1);
    long r2 = tile >> 1;
    int f3 = (int)(r2 % FD); r2 /= FD;
    int f2c = (int)(r2 % FD); r2 /= FD;
    int f1 = (int)(r2 % FD); r2 /= FD;
    int b  = (int)r2;
    if (b >= NB) return;

    int half = lane >> 4, lm = lane & 15;
    int n0 = f4t * 16;
    int f4 = n0 + lm;

    v8f acc = {};
    int nChunks = K / 32;                            // Ci * 3
    for (int c = 0; c < nChunks; ++c) {
        int kbase = c * 32;
        // A tile from LDS in contiguous k order (k = 16*half + s)
        const __bf16* ap = aLds + lm * K + kbase + half * 16;
        v16bf a = *(const v16bf*)(ap);
        // B tile: gather 16 tap values for column n0+lm
        int ci = c / 3;
        int tapbase = (c % 3) * 32 + half * 16;
        const float* xb = x + ((size_t)b * Ci + ci) * (size_t)POS4;
        v16bf bv;
        #pragma unroll
        for (int t = 0; t < 16; ++t) {
            int tap = tapbase + t;
            float val = 0.0f;
            if (tap < 81) {
                int i1 = tap / 27, rm = tap % 27;
                int i2 = rm / 9, i3 = (rm % 9) / 3, i4 = rm % 3;
                int p1 = f1 + i1 - 1, p2 = f2c + i2 - 1;
                int p3 = f3 + i3 - 1, p4 = f4 + i4 - 1;
                if ((unsigned)p1 < FD && (unsigned)p2 < FD &&
                    (unsigned)p3 < FD && (unsigned)p4 < FD)
                    val = xb[(((size_t)p1 * FD + p2) * FD + p3) * FD + p4];
            }
            bv[t] = f2bf(val);
        }
        acc = __builtin_amdgcn_wmma_f32_16x16x32_bf16(
                  false, a, false, bv, (short)0, acc, false, false);
    }
    // epilogue: bias + relu, scatter rows (Co may be < 16)
    #pragma unroll
    for (int rr = 0; rr < 8; ++rr) {
        int co = rr + half * 8;
        if (co < Co) {
            size_t idx = ((size_t)b * Co + co) * (size_t)POS4
                       + ((((size_t)f1 * FD + f2c) * FD + f3) * FD + f4);
            float v = acc[rr] + bias[co];
            y[idx] = v > 0.f ? v : 0.f;
        }
    }
}

// ---------------------------------------------------------------------------
// 7) Symmetric combine: out[b,i,j] = ya[b,i,j] + yb[b,j,i]
// ---------------------------------------------------------------------------
__global__ void combine_kernel(const float* __restrict__ ya,
                               const float* __restrict__ yb,
                               float* __restrict__ out) {
    size_t idx = (size_t)blockIdx.x * blockDim.x + threadIdx.x;
    if (idx >= ((size_t)NB << 20)) return;
    int b = (int)(idx >> 20), i = (int)((idx >> 10) & 1023), j = (int)(idx & 1023);
    out[idx] = ya[idx] + yb[((size_t)b << 20) + ((size_t)j << 10) + i];
}

// ---------------------------------------------------------------------------
extern "C" void kernel_launch(void* const* d_in, const int* in_sizes, int n_in,
                              void* d_out, int out_size, void* d_ws, size_t ws_size,
                              hipStream_t stream) {
    const float* fA = (const float*)d_in[0];
    const float* fB = (const float*)d_in[1];
    const float* w1 = (const float*)d_in[2];
    const float* b1 = (const float*)d_in[3];
    const float* w2 = (const float*)d_in[4];
    const float* b2 = (const float*)d_in[5];
    const float* w3 = (const float*)d_in[6];
    const float* b3 = (const float*)d_in[7];
    float* out = (float*)d_out;

    char* ws = (char*)d_ws;
    size_t off = 0;
    auto alloc = [&](size_t bytes) { void* p = ws + off; off = (off + bytes + 255) & ~(size_t)255; return p; };

    const size_t corrElems = (size_t)NB << 20;                 // 4M floats
    __bf16* faT  = (__bf16*)alloc((size_t)NB * NPOS * 256 * 2);
    __bf16* fbT  = (__bf16*)alloc((size_t)NB * NPOS * 256 * 2);
    float* corr  = (float*)alloc(corrElems * 4);
    float* maxA  = (float*)alloc((size_t)NB * NPOS * 4);
    float* maxB  = (float*)alloc((size_t)NB * NPOS * 4);
    float* x0    = (float*)alloc(corrElems * 4);
    float* xT    = (float*)alloc(corrElems * 4);
    float* bufA  = (float*)alloc((size_t)NB * 10 * POS4 * 4);  // layer intermediates
    float* bufB  = (float*)alloc((size_t)NB * 10 * POS4 * 4);
    float* y3a   = (float*)alloc(corrElems * 4);
    float* y3b   = (float*)alloc(corrElems * 4);
    float* pre   = (float*)alloc(corrElems * 4);

    const int ewBlocks = (int)((corrElems + 255) / 256);

    // 1) normalize + transpose to bf16
    l2norm_transpose_kernel<<<(NB * NPOS + 255) / 256, 256, 0, stream>>>(fA, faT, 256, NPOS);
    l2norm_transpose_kernel<<<(NB * NPOS + 255) / 256, 256, 0, stream>>>(fB, fbT, 256, NPOS);

    // 2) correlation GEMM (WMMA) with relu + l2norm epilogue
    corr_wmma_kernel<<<2048, 256, 0, stream>>>(faT, fbT, corr);

    // 3) mutual matching #1
    rowmax_kernel<<<NB * NPOS, 256, 0, stream>>>(corr, maxA, NPOS);
    colmax_kernel<<<(NB * NPOS + 255) / 256, 256, 0, stream>>>(corr, maxB);
    mutual_match_kernel<<<ewBlocks, 256, 0, stream>>>(corr, maxA, maxB, x0);

    // 4) symmetric NC stack: path A on x0, path B on x0^T
    transpose_kernel<<<ewBlocks, 256, 0, stream>>>(x0, xT);

    const int convBlocks = (NB * FD * FD * FD * 2) / 8;        // 8 waves/block
    auto shBytes = [](int Ci, int Co) -> size_t {
        return (size_t)((16 * Ci * 96 * 2 + 15) & ~15) + (size_t)Co * Ci * 81 * 4;
    };

    conv4d_wmma_kernel<<<convBlocks, 256, shBytes(1, 10),  stream>>>(x0,   w1, b1, bufA, 1, 10);
    conv4d_wmma_kernel<<<convBlocks, 256, shBytes(10, 10), stream>>>(bufA, w2, b2, bufB, 10, 10);
    conv4d_wmma_kernel<<<convBlocks, 256, shBytes(10, 1),  stream>>>(bufB, w3, b3, y3a, 10, 1);

    conv4d_wmma_kernel<<<convBlocks, 256, shBytes(1, 10),  stream>>>(xT,   w1, b1, bufA, 1, 10);
    conv4d_wmma_kernel<<<convBlocks, 256, shBytes(10, 10), stream>>>(bufA, w2, b2, bufB, 10, 10);
    conv4d_wmma_kernel<<<convBlocks, 256, shBytes(10, 1),  stream>>>(bufB, w3, b3, y3b, 10, 1);

    combine_kernel<<<ewBlocks, 256, 0, stream>>>(y3a, y3b, pre);

    // 5) mutual matching #2 -> d_out
    rowmax_kernel<<<NB * NPOS, 256, 0, stream>>>(pre, maxA, NPOS);
    colmax_kernel<<<(NB * NPOS + 255) / 256, 256, 0, stream>>>(pre, maxB);
    mutual_match_kernel<<<ewBlocks, 256, 0, stream>>>(pre, maxA, maxB, out);
}
